// MRGCN_43069932045115
// MI455X (gfx1250) — matrix operand order repaired and checked by reading, not compiled, not run on hardware
//
#include <hip/hip_runtime.h>

// Problem constants (match reference)
#define NN_  4096
#define RR_  4
#define FIN_ 128
#define HH_  64
#define CC_  16
#define TT_  512

typedef _Float16 v4h  __attribute__((ext_vector_type(4)));
typedef _Float16 v8h  __attribute__((ext_vector_type(8)));
typedef _Float16 v16h __attribute__((ext_vector_type(16)));
typedef float    v8f  __attribute__((ext_vector_type(8)));

static __device__ __forceinline__ v16h pack16(v8h lo, v8h hi) {
    union { v16h v; v8h h[2]; } u;
    u.h[0] = lo; u.h[1] = hi;
    return u.v;
}

// ---------------------------------------------------------------------------
// K0: d[r][n] = rsqrt(max(1, rowsum(A with diag forced to 1)))
// ---------------------------------------------------------------------------
__global__ __launch_bounds__(256) void k_degree(const float* __restrict__ A,
                                                float* __restrict__ d) {
    const int lane = threadIdx.x & 31;
    const int wave = threadIdx.x >> 5;
    const int rowg = blockIdx.x * 8 + wave;       // 0 .. R*N-1
    const int n = rowg & (NN_ - 1);
    const float* arow = A + (long)rowg * NN_;
    float s = 0.f;
    for (int j = lane; j < NN_; j += 32) s += arow[j];
    for (int off = 16; off; off >>= 1) s += __shfl_xor(s, off, 32);
    if (lane == 0) {
        float deg = s - arow[n] + 1.0f;           // force self-loop = 1
        deg = fmaxf(deg, 1.0f);
        d[rowg] = rsqrtf(deg);
    }
}

// ---------------------------------------------------------------------------
// K1: An16[r][i][j] = f16( d_i * d_j * (i==j ? 1 : A_ij) )   (4 elems/thread)
// ---------------------------------------------------------------------------
__global__ __launch_bounds__(256) void k_an16(const float* __restrict__ A,
                                              const float* __restrict__ d,
                                              _Float16* __restrict__ An16) {
    const long e = ((long)blockIdx.x * blockDim.x + threadIdx.x) * 4;
    const int  r   = (int)(e >> 24);              // N*N = 2^24
    const long rem = e & ((1L << 24) - 1);
    const int  i   = (int)(rem >> 12);
    const int  j0  = (int)(rem & (NN_ - 1));
    const float4 a = *(const float4*)(A + e);
    const float di = d[r * NN_ + i];
    const float* dj = d + r * NN_ + j0;
    float va[4] = {a.x, a.y, a.z, a.w};
    v4h o;
#pragma unroll
    for (int t = 0; t < 4; ++t) {
        float v = (i == j0 + t) ? 1.0f : va[t];
        o[t] = (_Float16)(v * di * dj[t]);
    }
    *(v4h*)(An16 + e) = o;                        // 8-byte store
}

// ---------------------------------------------------------------------------
// Big WMMA kernel: OUT[y][4096][64] f16 = act( (S-combined A f16) @ B + bias )
//   A combine signs are template params -> v_pk_add_f16 with neg modifiers.
//   B panel [64][4096] f16 staged to LDS in 512-K chunks with
//   global_load_async_to_lds_b128 (ASYNCcnt), double buffered; fragments read
//   with padded, conflict-free ds_load_b128. A stream register-pipelined +
//   far global_prefetch so demand loads hit GL2.
// ---------------------------------------------------------------------------
template <int NMAT, int S1, int S2, int S3>
__global__ __launch_bounds__(256) void k_bigmm(
    const _Float16* __restrict__ A0, const _Float16* __restrict__ A1,
    const _Float16* __restrict__ A2, const _Float16* __restrict__ A3,
    long aStride,
    const _Float16* __restrict__ Bt, long bStride,
    const float* __restrict__ bias, int biasStride, int do_relu,
    _Float16* __restrict__ out, long oStride) {
    constexpr int CH = 512;               // K elements per LDS chunk
    constexpr int LROW = CH + 8;          // padded row (halves): bank-conflict free
    constexpr int NCHUNK = NN_ / CH;
    extern __shared__ _Float16 ldsb[];    // 2 * 64 * LROW halves

    const int y = blockIdx.y;
    A0 += (long)y * aStride;
    if (NMAT > 1 && A1) A1 += (long)y * aStride;
    if (NMAT > 2 && A2) A2 += (long)y * aStride;
    if (NMAT > 3 && A3) A3 += (long)y * aStride;
    Bt  += (long)y * bStride;
    out += (long)y * oStride;
    const float* bvec = bias ? bias + (long)y * biasStride : nullptr;

    const int lane = threadIdx.x & 31;
    const int wave = threadIdx.x >> 5;
    const int rowTile = (blockIdx.x * 8 + wave) * 16;
    const int m  = lane & 15;
    const int hi = lane >> 4;
    const long arow = (long)(rowTile + m) * NN_;

    v8f acc[4] = {v8f{}, v8f{}, v8f{}, v8f{}};

    // issue async copy of B[all 64 cols][kc .. kc+CH) into LDS buffer `buf`
    auto issueB = [&](int kc, int buf) {
        const int t = threadIdx.x;
#pragma unroll
        for (int i = 0; i < (64 * CH / 8) / 256; ++i) {   // 16 xfers/thread
            const int idx = t + 256 * i;
            const int row = idx >> 6;                     // 64 xfers per row
            const int seg = idx & 63;
            const _Float16* g = Bt + (long)row * NN_ + kc + seg * 8;
            const _Float16* l = &ldsb[(size_t)buf * 64 * LROW + row * LROW + seg * 8];
            asm volatile("global_load_async_to_lds_b128 %0, %1, off"
                         :: "v"((unsigned)(unsigned long long)l),
                            "v"((unsigned long long)g)
                         : "memory");
        }
    };

    auto loadAfrag = [&](int kg, v8h* a) {
        const int ka = kg + (hi << 3);                    // K [0..7] / [8..15]
        const int kb = ka + 16;                           // K [16..23] / [24..31]
        a[0] = *(const v8h*)(A0 + arow + ka);
        a[1] = *(const v8h*)(A0 + arow + kb);
        if (NMAT > 1) { a[2] = *(const v8h*)(A1 + arow + ka);
                        a[3] = *(const v8h*)(A1 + arow + kb); }
        if (NMAT > 2) { a[4] = *(const v8h*)(A2 + arow + ka);
                        a[5] = *(const v8h*)(A2 + arow + kb); }
        if (NMAT > 3) { a[6] = *(const v8h*)(A3 + arow + ka);
                        a[7] = *(const v8h*)(A3 + arow + kb); }
    };

    v8h aCur[2 * NMAT], aNxt[2 * NMAT];

    issueB(0, 0);
    asm volatile("s_wait_asynccnt 0" ::: "memory");
    __syncthreads();
    loadAfrag(0, aCur);

    int kg = 0;
    for (int c = 0; c < NCHUNK; ++c) {
        if (c + 1 < NCHUNK) issueB((c + 1) * CH, (c + 1) & 1);
        const _Float16* L = ldsb + (size_t)(c & 1) * 64 * LROW;
        for (int kk = 0; kk < CH; kk += 32) {
            const int kgn = (kg + 32 < NN_) ? kg + 32 : kg;   // last iter: dup
            loadAfrag(kgn, aNxt);                              // pipeline A
            __builtin_prefetch(A0 + arow + kg + 2048, 0, 0);   // far GL2 prefetch
            if (NMAT > 1) __builtin_prefetch(A1 + arow + kg + 2048, 0, 0);
            if (NMAT > 2) __builtin_prefetch(A2 + arow + kg + 2048, 0, 0);
            if (NMAT > 3) __builtin_prefetch(A3 + arow + kg + 2048, 0, 0);
            // compile-time sign combine (v_pk_add_f16 w/ neg modifiers)
            v8h lo = aCur[0], hh = aCur[1];
            if (NMAT > 1) { if (S1 > 0) { lo += aCur[2]; hh += aCur[3]; }
                            else        { lo -= aCur[2]; hh -= aCur[3]; } }
            if (NMAT > 2) { if (S2 > 0) { lo += aCur[4]; hh += aCur[5]; }
                            else        { lo -= aCur[4]; hh -= aCur[5]; } }
            if (NMAT > 3) { if (S3 > 0) { lo += aCur[6]; hh += aCur[7]; }
                            else        { lo -= aCur[6]; hh -= aCur[7]; } }
            const v16h af = pack16(lo, hh);
            const int kb0 = kk + (hi << 3);               // K [0..7]/[8..15] then +16
#pragma unroll
            for (int j = 0; j < 4; ++j) {
                const _Float16* bp = L + (j * 16 + m) * LROW + kb0;
                const v16h bf = pack16(*(const v8h*)bp, *(const v8h*)(bp + 16));
                acc[j] = __builtin_amdgcn_wmma_f32_16x16x32_f16(
                    false, af, false, bf, (short)0, acc[j], false, false);
            }
#pragma unroll
            for (int q = 0; q < 2 * NMAT; ++q) aCur[q] = aNxt[q];
            kg += 32;
        }
        asm volatile("s_wait_asynccnt 0" ::: "memory");    // next buf copied
        __syncthreads();                                   // all waves past reads
    }

    // Epilogue per 16x16 C layout: lane holds rows (hi*8..hi*8+7), col = m
#pragma unroll
    for (int j = 0; j < 4; ++j) {
        const int col = j * 16 + m;
        const float b = bvec ? bvec[col] : 0.0f;
#pragma unroll
        for (int v = 0; v < 8; ++v) {
            float x = acc[j][v] + b;
            if (do_relu) x = fmaxf(x, 0.0f);
            out[(long)(rowTile + (hi << 3) + v) * HH_ + col] = (_Float16)x;
        }
    }
}

// ---------------------------------------------------------------------------
// Small GEMM: out = A[4096][K] @ W[K][64]   (thin matmuls, VALU)
// ---------------------------------------------------------------------------
template <typename TA, typename TO, bool TRANS>
__global__ __launch_bounds__(256) void k_gemm_small(const TA* __restrict__ A,
                                                    int lda, int K,
                                                    const float* __restrict__ W,
                                                    TO* __restrict__ out) {
    const int n = blockIdx.x * blockDim.y + threadIdx.y;
    const int c = threadIdx.x;                    // 0..63
    const TA* a = A + (size_t)n * lda;
    float acc = 0.f;
    for (int k = 0; k < K; ++k) acc += (float)a[k] * W[k * HH_ + c];
    if (TRANS) out[(size_t)c * NN_ + n] = (TO)acc;
    else       out[(size_t)n * HH_ + c] = (TO)acc;
}

// ---------------------------------------------------------------------------
// Freq real-parts of mrgco_w[l]: Wf[k][f][c], k in {0,1,2}
// ---------------------------------------------------------------------------
__global__ __launch_bounds__(256) void k_wfreq(const float* __restrict__ mw,
                                               float* __restrict__ Wf) {
    const int idx = blockIdx.x * blockDim.x + threadIdx.x; // k*4096 + f*64 + c
    if (idx >= 3 * HH_ * HH_) return;
    const int k = idx >> 12;
    const int fc = idx & 4095;
    const float* p = mw + fc * RR_;
    float r0 = p[0], r1 = p[1], r2 = p[2], r3 = p[3];
    Wf[idx] = (k == 0) ? (r0 + r1 + r2 + r3) : (k == 1) ? (r0 - r2)
                                                        : (r0 - r1 + r2 - r3);
}

// ---------------------------------------------------------------------------
// Freq real-parts of h2 ([4][N][64] f16) -> Xft [3][64][4096] (B-transposed)
// ---------------------------------------------------------------------------
__global__ __launch_bounds__(256) void k_xfreq(const _Float16* __restrict__ h2,
                                               _Float16* __restrict__ Xft) {
    const int idx = blockIdx.x * blockDim.x + threadIdx.x;  // n*64 + h
    const int n = idx >> 6, h = idx & 63;
    const float t0 = (float)h2[0 * NN_ * HH_ + idx];
    const float t1 = (float)h2[1 * NN_ * HH_ + idx];
    const float t2 = (float)h2[2 * NN_ * HH_ + idx];
    const float t3 = (float)h2[3 * NN_ * HH_ + idx];
    const size_t o = (size_t)h * NN_ + n;
    Xft[0 * HH_ * NN_ + o] = (_Float16)(t0 + t1 + t2 + t3);
    Xft[1 * HH_ * NN_ + o] = (_Float16)(t0 - t2);
    Xft[2 * HH_ * NN_ + o] = (_Float16)(t0 - t1 + t2 - t3);
}

// ---------------------------------------------------------------------------
// TP2 epilogue: ifft (E3==E1) + /4 + relu; emit next Xft and/or flat
// ---------------------------------------------------------------------------
__global__ __launch_bounds__(256) void k_tp2_epi(const float* __restrict__ E,
                                                 _Float16* __restrict__ Xft_next,
                                                 float* __restrict__ flat) {
    const int idx = blockIdx.x * blockDim.x + threadIdx.x;  // n*64 + h
    const int n = idx >> 6, h = idx & 63;
    const float e0 = E[idx];
    const float e1 = E[NN_ * HH_ + idx];
    const float e2 = E[2 * NN_ * HH_ + idx];
    const float t0 = fmaxf((e0 + 2.f * e1 + e2) * 0.25f, 0.f);
    const float t1 = fmaxf((e0 - e2) * 0.25f, 0.f);
    const float t2 = fmaxf((e0 - 2.f * e1 + e2) * 0.25f, 0.f);
    const float t3 = t1;
    if (Xft_next) {
        const size_t o = (size_t)h * NN_ + n;
        Xft_next[0 * HH_ * NN_ + o] = (_Float16)(t0 + t1 + t2 + t3);
        Xft_next[1 * HH_ * NN_ + o] = (_Float16)(t0 - t2);
        Xft_next[2 * HH_ * NN_ + o] = (_Float16)(t0 - t1 + t2 - t3);
    }
    if (flat) {
        float* fr = flat + (size_t)n * (HH_ * RR_) + h * RR_;
        fr[0] = t0; fr[1] = t1; fr[2] = t2; fr[3] = t3;
    }
}

// ---------------------------------------------------------------------------
// Head: per target row -> lin1+relu -> lin2 -> logits + per-sample NLL
// ---------------------------------------------------------------------------
__global__ __launch_bounds__(64) void k_head(const float* __restrict__ flat,
                                             const int* __restrict__ tx,
                                             const int* __restrict__ tgt,
                                             const float* __restrict__ w1,
                                             const float* __restrict__ b1,
                                             const float* __restrict__ w2,
                                             const float* __restrict__ b2,
                                             float* __restrict__ y_out,
                                             float* __restrict__ nll) {
    __shared__ float z[HH_];
    __shared__ float yb[CC_];
    const int t = blockIdx.x;
    const int c = threadIdx.x;
    const int row = tx[t];
    const float* fr = flat + (size_t)row * (HH_ * RR_);
    float acc = b1[c];
    for (int k = 0; k < HH_ * RR_; ++k) acc += fr[k] * w1[k * HH_ + c];
    z[c] = fmaxf(acc, 0.f);
    __syncthreads();
    if (c < CC_) {
        float a = b2[c];
        for (int k = 0; k < HH_; ++k) a += z[k] * w2[k * CC_ + c];
        yb[c] = a;
        y_out[t * CC_ + c] = a;
    }
    __syncthreads();
    if (c == 0) {
        float mx = -1e30f;
        for (int k = 0; k < CC_; ++k) mx = fmaxf(mx, yb[k]);
        float s = 0.f;
        for (int k = 0; k < CC_; ++k) s += expf(yb[k] - mx);
        nll[t] = -(yb[tgt[t]] - mx - logf(s));
    }
}

// Deterministic loss reduction (fixed-order tree, no float atomics)
__global__ __launch_bounds__(256) void k_loss(const float* __restrict__ nll,
                                              float* __restrict__ out) {
    __shared__ float s[256];
    const int i = threadIdx.x;
    s[i] = nll[i] + nll[i + 256];
    __syncthreads();
    for (int off = 128; off; off >>= 1) {
        if (i < off) s[i] += s[i + off];
        __syncthreads();
    }
    if (i == 0) out[0] = s[0] / (float)TT_;
}

// ---------------------------------------------------------------------------
extern "C" void kernel_launch(void* const* d_in, const int* in_sizes, int n_in,
                              void* d_out, int out_size, void* d_ws, size_t ws_size,
                              hipStream_t stream) {
    (void)in_sizes; (void)n_in; (void)out_size; (void)ws_size;
    const float* A   = (const float*)d_in[0];
    const float* X   = (const float*)d_in[1];
    const int*   tx  = (const int*)d_in[2];
    const int*   tgt = (const int*)d_in[3];
    const float* w1  = (const float*)d_in[4];
    const float* b1  = (const float*)d_in[5];
    const float* w2  = (const float*)d_in[6];
    const float* b2  = (const float*)d_in[7];
    const float* mw  = (const float*)d_in[8];
    const float* l1w = (const float*)d_in[9];
    const float* l1b = (const float*)d_in[10];
    const float* l2w = (const float*)d_in[11];
    const float* l2b = (const float*)d_in[12];
    float* outp = (float*)d_out;

    // workspace carve (256B aligned)
    char* w = (char*)d_ws;
    auto carve = [&](size_t bytes) {
        char* p = w;
        w += (bytes + 255) & ~(size_t)255;
        return p;
    };
    const size_t NNe = (size_t)NN_ * NN_;
    float*     d_deg = (float*)carve((size_t)RR_ * NN_ * 4);
    _Float16*  An16  = (_Float16*)carve((size_t)RR_ * NNe * 2);        // 134 MB
    _Float16*  Bt    = (_Float16*)carve((size_t)RR_ * HH_ * NN_ * 2);
    _Float16*  h16   = (_Float16*)carve((size_t)RR_ * NN_ * HH_ * 2);
    _Float16*  Xft   = (_Float16*)carve((size_t)3 * HH_ * NN_ * 2);
    _Float16*  C16   = (_Float16*)carve((size_t)3 * NN_ * HH_ * 2);
    float*     E     = (float*)carve((size_t)3 * NN_ * HH_ * 4);
    float*     Wf    = (float*)carve((size_t)3 * HH_ * HH_ * 4);
    float*     flat  = (float*)carve((size_t)NN_ * HH_ * RR_ * 4);
    float*     nll   = (float*)carve((size_t)TT_ * 4);

    const dim3 gblk(64, 4);
    const size_t SHB = 2ull * 64 * (512 + 8) * 2;   // 133,120 B dyn LDS

    // --- normalization ---
    k_degree<<<RR_ * NN_ / 8, 256, 0, stream>>>(A, d_deg);
    k_an16<<<(unsigned)(RR_ * NNe / 4 / 256), 256, 0, stream>>>(A, d_deg, An16);

    // --- GCN layer 1: h1 = relu(An @ (X@W1) + b1), all relations in one grid ---
    for (int r = 0; r < RR_; ++r)
        k_gemm_small<float, _Float16, true><<<NN_ / 4, gblk, 0, stream>>>(
            X, FIN_, FIN_, w1 + (size_t)r * FIN_ * HH_, Bt + (size_t)r * HH_ * NN_);
    k_bigmm<1, 0, 0, 0><<<dim3(32, RR_), 256, SHB, stream>>>(
        An16, nullptr, nullptr, nullptr, (long)NNe,
        Bt, (long)HH_ * NN_, b1, HH_, 1, h16, (long)NN_ * HH_);

    // --- GCN layer 2: h2 = relu(An @ (h1@W2) + b2) ---
    for (int r = 0; r < RR_; ++r)
        k_gemm_small<_Float16, _Float16, true><<<NN_ / 4, gblk, 0, stream>>>(
            h16 + (size_t)r * NN_ * HH_, HH_, HH_, w2 + (size_t)r * HH_ * HH_,
            Bt + (size_t)r * HH_ * NN_);
    k_bigmm<1, 0, 0, 0><<<dim3(32, RR_), 256, SHB, stream>>>(
        An16, nullptr, nullptr, nullptr, (long)NNe,
        Bt, (long)HH_ * NN_, b2, HH_, 1, h16, (long)NN_ * HH_);

    // --- to frequency domain ---
    k_xfreq<<<NN_ * HH_ / 256, 256, 0, stream>>>(h16, Xft);

    // --- MRGCO layers (freq-domain; fft∘ifft identity inside a layer) ---
    for (int l = 0; l < 2; ++l) {
        k_wfreq<<<48, 256, 0, stream>>>(mw + (size_t)l * HH_ * HH_ * RR_, Wf);
        // TP1: C_k = Af_k @ Xf_k (A relations combined in f16 registers)
        k_bigmm<4, 1, 1, 1><<<dim3(32, 1), 256, SHB, stream>>>(     // freq 0: ++++
            An16, An16 + NNe, An16 + 2 * NNe, An16 + 3 * NNe, 0,
            Xft + 0 * (size_t)HH_ * NN_, 0, nullptr, 0, 0,
            C16 + 0 * (size_t)NN_ * HH_, 0);
        k_bigmm<2, -1, 0, 0><<<dim3(32, 1), 256, SHB, stream>>>(    // freq 1: r0-r2
            An16, An16 + 2 * NNe, nullptr, nullptr, 0,
            Xft + 1 * (size_t)HH_ * NN_, 0, nullptr, 0, 0,
            C16 + 1 * (size_t)NN_ * HH_, 0);
        k_bigmm<4, -1, 1, -1><<<dim3(32, 1), 256, SHB, stream>>>(   // freq 2: +-+-
            An16, An16 + NNe, An16 + 2 * NNe, An16 + 3 * NNe, 0,
            Xft + 2 * (size_t)HH_ * NN_, 0, nullptr, 0, 0,
            C16 + 2 * (size_t)NN_ * HH_, 0);
        // TP2: E_k = C_k @ Wf_k
        for (int k = 0; k < 3; ++k)
            k_gemm_small<_Float16, float, false><<<NN_ / 4, gblk, 0, stream>>>(
                C16 + (size_t)k * NN_ * HH_, HH_, HH_, Wf + (size_t)k * HH_ * HH_,
                E + (size_t)k * NN_ * HH_);
        // ifft + relu; next-layer fft or final flatten
        k_tp2_epi<<<NN_ * HH_ / 256, 256, 0, stream>>>(
            E, (l == 0) ? Xft : nullptr, (l == 1) ? flat : nullptr);
    }

    // --- head (only target rows) + deterministic loss ---
    k_head<<<TT_, 64, 0, stream>>>(flat, tx, tgt, l1w, l1b, l2w, l2b, outp + 1, nll);
    k_loss<<<1, 256, 0, stream>>>(nll, outp);
}
